// ReadingOrderModel_62027917689265
// MI455X (gfx1250) — compile-verified, matrix operand order checked by reading, not run on hardware
//
#include <hip/hip_runtime.h>
#include <hip/hip_bf16.h>
#include <math.h>

// ---------------------------------------------------------------------------
// Types for CDNA5 WMMA (wave32): v_wmma_f32_16x16x32_bf16
// ---------------------------------------------------------------------------
typedef __attribute__((ext_vector_type(16))) __bf16 bf16x16;
typedef __attribute__((ext_vector_type(8)))  __bf16 bf16x8;
typedef __attribute__((ext_vector_type(8)))  float  f32x8;

union Frag16 { bf16x16 v; bf16x8 h[2]; };

__device__ __forceinline__ float wave_sum(float v) {
#pragma unroll
  for (int o = 16; o > 0; o >>= 1) v += __shfl_xor(v, o, 32);
  return v;
}
__device__ __forceinline__ float wave_max(float v) {
#pragma unroll
  for (int o = 16; o > 0; o >>= 1) v = fmaxf(v, __shfl_xor(v, o, 32));
  return v;
}

// ---------------------------------------------------------------------------
// Batched GEMM: C[z] = act( A[z][M,K] * B' + bias + residual )
//   TRANS_B=1: B given as W[Ncols,K] row-major (C = A @ W^T)  -- projections, Q@K^T
//   TRANS_B=0: B given as V[K,Ncols] row-major (C = A @ V)    -- attn @ V
// Compile-time HAS_BIAS / HAS_RES / RELU keep the epilogue branch-free.
// Requires M % 64 == 0, Ncols % 64 == 0, K % 64 == 0 (true for all calls here).
// Block: 128 threads = 4 waves (2x2); wave owns 32x32 out tile; K-step 64 ->
// 8 v_wmma per wave per barrier. Double-buffered LDS: global loads for tile
// t+1 are in flight while tile t runs on the matrix pipe; 1 barrier/tile.
// ---------------------------------------------------------------------------
#define TILE_M 64
#define TILE_N 64
#define TILE_K 64
#define LDS_ST 72  // 64 + 8 pad bf16 per row (144B, 16B aligned)

template <int TRANS_B, int HAS_BIAS, int HAS_RES, int RELU>
__global__ __launch_bounds__(128) void gemm_wmma(
    const float* __restrict__ A, int lda, long long bsA,
    const float* __restrict__ B, int ldb, long long bsB,
    float* __restrict__ C, int ldc, long long bsC,
    const float* __restrict__ bias,
    const float* __restrict__ residual, int ldr,
    int M, int Ncols, int K) {
  __shared__ __bf16 lsA[2][TILE_M * LDS_ST];
  __shared__ __bf16 lsB[2][TILE_N * LDS_ST];

  const int tid  = threadIdx.x;
  const int lane = tid & 31;
  const int wid  = tid >> 5;
  const int wr   = wid >> 1;   // wave row (0..1)
  const int wc   = wid & 1;    // wave col (0..1)
  const int l15  = lane & 15;
  const int lh   = lane >> 4;  // lane half (0/1)
  const int bm0  = blockIdx.y * TILE_M;
  const int bn0  = blockIdx.x * TILE_N;

  A += (long long)blockIdx.z * bsA;
  B += (long long)blockIdx.z * bsB;
  C += (long long)blockIdx.z * bsC;

  f32x8 acc[2][2] = {};
  float4 ra[8], rb[8];

  // issue global loads for K-tile at k0 into registers (kept in flight)
  auto load_regs = [&](int k0) {
#pragma unroll
    for (int i = 0; i < 8; ++i) {
      const int c4  = tid + i * 128;   // float4 chunk id in [0,1024)
      const int row = c4 >> 4;         // 16 float4 per 64-float row
      const int kq  = (c4 & 15) << 2;
      ra[i] = *(const float4*)(A + (size_t)(bm0 + row) * lda + k0 + kq);
    }
    if (TRANS_B) {
#pragma unroll
      for (int i = 0; i < 8; ++i) {
        const int c4  = tid + i * 128;
        const int row = c4 >> 4;
        const int kq  = (c4 & 15) << 2;
        rb[i] = *(const float4*)(B + (size_t)(bn0 + row) * ldb + k0 + kq);
      }
    } else {
      // k-major float4 loads (coalesced); scattered to LDS columns on store
#pragma unroll
      for (int i = 0; i < 8; ++i) {
        const int c4 = tid + i * 128;
        const int kk = c4 >> 4;
        const int nq = (c4 & 15) << 2;
        rb[i] = *(const float4*)(B + (size_t)(k0 + kk) * ldb + bn0 + nq);
      }
    }
  };

  // convert registers -> bf16 LDS tiles (lsA[m][k], lsB[n][k])
  auto store_tile = [&](int buf) {
#pragma unroll
    for (int i = 0; i < 8; ++i) {
      const int c4  = tid + i * 128;
      const int row = c4 >> 4;
      const int kq  = (c4 & 15) << 2;
      __bf16* dst = &lsA[buf][row * LDS_ST + kq];
      dst[0] = (__bf16)ra[i].x; dst[1] = (__bf16)ra[i].y;
      dst[2] = (__bf16)ra[i].z; dst[3] = (__bf16)ra[i].w;
    }
    if (TRANS_B) {
#pragma unroll
      for (int i = 0; i < 8; ++i) {
        const int c4  = tid + i * 128;
        const int row = c4 >> 4;
        const int kq  = (c4 & 15) << 2;
        __bf16* dst = &lsB[buf][row * LDS_ST + kq];
        dst[0] = (__bf16)rb[i].x; dst[1] = (__bf16)rb[i].y;
        dst[2] = (__bf16)rb[i].z; dst[3] = (__bf16)rb[i].w;
      }
    } else {
#pragma unroll
      for (int i = 0; i < 8; ++i) {
        const int c4 = tid + i * 128;
        const int kk = c4 >> 4;
        const int nq = (c4 & 15) << 2;
        lsB[buf][(nq + 0) * LDS_ST + kk] = (__bf16)rb[i].x;
        lsB[buf][(nq + 1) * LDS_ST + kk] = (__bf16)rb[i].y;
        lsB[buf][(nq + 2) * LDS_ST + kk] = (__bf16)rb[i].z;
        lsB[buf][(nq + 3) * LDS_ST + kk] = (__bf16)rb[i].w;
      }
    }
  };

  // 2 K-substeps of 32; fragment loads per ISA 16-bit A 16x32 layout:
  // lanes 0-15: K {0..7,16..23}; lanes 16-31: K {8..15,24..31}
  auto compute = [&](int buf) {
#pragma unroll
    for (int ks = 0; ks < TILE_K; ks += 32) {
      Frag16 aF[2], bF[2];
#pragma unroll
      for (int f = 0; f < 2; ++f) {
        const __bf16* pa = &lsA[buf][(wr * 32 + f * 16 + l15) * LDS_ST + ks];
        aF[f].h[0] = *(const bf16x8*)(pa + lh * 8);
        aF[f].h[1] = *(const bf16x8*)(pa + 16 + lh * 8);
        const __bf16* pb = &lsB[buf][(wc * 32 + f * 16 + l15) * LDS_ST + ks];
        bF[f].h[0] = *(const bf16x8*)(pb + lh * 8);
        bF[f].h[1] = *(const bf16x8*)(pb + 16 + lh * 8);
      }
#pragma unroll
      for (int fm = 0; fm < 2; ++fm)
#pragma unroll
        for (int fn = 0; fn < 2; ++fn)
          acc[fm][fn] = __builtin_amdgcn_wmma_f32_16x16x32_bf16(
              false, aF[fm].v, false, bF[fn].v, (short)0, acc[fm][fn], false, false);
    }
  };

  const int nk = K / TILE_K;
  load_regs(0);
  store_tile(0);
  __syncthreads();
  for (int t = 0; t < nk; ++t) {
    const int cur = t & 1;
    if (t + 1 < nk) load_regs((t + 1) * TILE_K);  // overlap with matrix pipe
    compute(cur);
    if (t + 1 < nk) {
      store_tile(cur ^ 1);
      __syncthreads();
    }
  }

  // --- epilogue: C/D layout (VGPR r, lane) -> M = r + 8*(lane>=16), N = lane&15 ---
#pragma unroll
  for (int fm = 0; fm < 2; ++fm) {
    const int row0 = bm0 + wr * 32 + fm * 16 + lh * 8;
#pragma unroll
    for (int fn = 0; fn < 2; ++fn) {
      const int col = bn0 + wc * 32 + fn * 16 + l15;
      float bv = 0.0f;
      if (HAS_BIAS) bv = bias[col];
      float* cp = C + (size_t)row0 * ldc + col;
      const float* rp = HAS_RES ? (residual + (size_t)row0 * ldr + col) : cp;
#pragma unroll
      for (int r = 0; r < 8; ++r) {
        float v = acc[fm][fn][r];
        if (HAS_BIAS) v += bv;
        if (HAS_RES) v += *rp;
        if (RELU) v = fmaxf(v, 0.0f);
        *cp = v;
        cp += ldc;
        if (HAS_RES) rp += ldr;
      }
    }
  }
}

// ---------------------------------------------------------------------------
// Row softmax over `len` elements (len % 128 == 0, len/128 <= 4). scale folded in.
// ---------------------------------------------------------------------------
__global__ __launch_bounds__(128) void softmax_rows(float* __restrict__ S, int len,
                                                    float scale) {
  float* row = S + (size_t)blockIdx.x * len;
  const int tid = threadIdx.x, lane = tid & 31, wid = tid >> 5;
  __shared__ float sh[4];
  const int nit = len >> 7;
  float v[4];
  float m = -3.0e38f;
  for (int i = 0; i < nit; ++i) { v[i] = row[tid + i * 128] * scale; m = fmaxf(m, v[i]); }
  m = wave_max(m);
  if (lane == 0) sh[wid] = m;
  __syncthreads();
  m = fmaxf(fmaxf(sh[0], sh[1]), fmaxf(sh[2], sh[3]));
  __syncthreads();
  float s = 0.0f;
  for (int i = 0; i < nit; ++i) { v[i] = expf(v[i] - m); s += v[i]; }
  s = wave_sum(s);
  if (lane == 0) sh[wid] = s;
  __syncthreads();
  const float inv = 1.0f / (sh[0] + sh[1] + sh[2] + sh[3]);
  for (int i = 0; i < nit; ++i) row[tid + i * 128] = v[i] * inv;
}

// ---------------------------------------------------------------------------
// LayerNorm: Y[row] = (X[row]-mean)*rsqrt(var+eps)*g + b
// ---------------------------------------------------------------------------
__global__ __launch_bounds__(256) void layernorm_k(const float* __restrict__ X,
                                                   float* __restrict__ Y,
                                                   const float* __restrict__ g,
                                                   const float* __restrict__ b, int d) {
  const float* x = X + (size_t)blockIdx.x * d;
  float* y = Y + (size_t)blockIdx.x * d;
  const int tid = threadIdx.x, lane = tid & 31, wid = tid >> 5;
  __shared__ float s1[8], s2[8];
  float sum = 0.0f, sq = 0.0f;
  for (int i = tid; i < d; i += 256) { const float v = x[i]; sum += v; sq += v * v; }
  sum = wave_sum(sum); sq = wave_sum(sq);
  if (lane == 0) { s1[wid] = sum; s2[wid] = sq; }
  __syncthreads();
  float t = 0.0f, q = 0.0f;
#pragma unroll
  for (int i = 0; i < 8; ++i) { t += s1[i]; q += s2[i]; }
  const float mean = t / d;
  const float var  = q / d - mean * mean;
  const float inv  = rsqrtf(var + 1e-5f);
  for (int i = tid; i < d; i += 256) y[i] = (x[i] - mean) * inv * g[i] + b[i];
}

// ---------------------------------------------------------------------------
// concat rows: o[i] = [t[i] | v[i]]
// ---------------------------------------------------------------------------
__global__ void concat_k(const float* __restrict__ t, const float* __restrict__ v,
                         float* __restrict__ o, int dt, int dv) {
  const int i = blockIdx.x;
  for (int j = threadIdx.x; j < dt; j += blockDim.x)
    o[(size_t)i * (dt + dv) + j] = t[(size_t)i * dt + j];
  for (int j = threadIdx.x; j < dv; j += blockDim.x)
    o[(size_t)i * (dt + dv) + dt + j] = v[(size_t)i * dv + j];
}

// ---------------------------------------------------------------------------
// Head: a[i] = x[i] @ lin_w[:, :D]^T + lin_b ; c[i] = x[i] @ lin_w[:, D:]^T
// lin_w is (2, 2D) row-major.
// ---------------------------------------------------------------------------
__global__ __launch_bounds__(256) void head_ac_k(const float* __restrict__ X,
                                                 const float* __restrict__ lw,
                                                 const float* __restrict__ lb,
                                                 float* __restrict__ a,
                                                 float* __restrict__ c, int D) {
  const int i = blockIdx.x;
  const float* x = X + (size_t)i * D;
  const int tid = threadIdx.x, lane = tid & 31, wid = tid >> 5;
  float p0 = 0, p1 = 0, p2 = 0, p3 = 0;
  for (int j = tid; j < D; j += 256) {
    const float v = x[j];
    p0 += v * lw[j];            // row0, cols [0,D)
    p1 += v * lw[2 * D + j];    // row1, cols [0,D)
    p2 += v * lw[D + j];        // row0, cols [D,2D)
    p3 += v * lw[3 * D + j];    // row1, cols [D,2D)
  }
  __shared__ float s[4][8];
  p0 = wave_sum(p0); p1 = wave_sum(p1); p2 = wave_sum(p2); p3 = wave_sum(p3);
  if (lane == 0) { s[0][wid] = p0; s[1][wid] = p1; s[2][wid] = p2; s[3][wid] = p3; }
  __syncthreads();
  if (tid == 0) {
    float r0 = 0, r1 = 0, r2 = 0, r3 = 0;
#pragma unroll
    for (int k = 0; k < 8; ++k) { r0 += s[0][k]; r1 += s[1][k]; r2 += s[2][k]; r3 += s[3][k]; }
    a[i * 2 + 0] = r0 + lb[0];
    a[i * 2 + 1] = r1 + lb[1];
    c[i * 2 + 0] = r2;
    c[i * 2 + 1] = r3;
  }
}

// ---------------------------------------------------------------------------
// probs / losses: one block per row i, thread j per column.
// ---------------------------------------------------------------------------
__global__ __launch_bounds__(384) void probs_k(
    const float* __restrict__ a, const float* __restrict__ c,
    const int* __restrict__ gt, float* __restrict__ out,
    float* __restrict__ cls_sum, float* __restrict__ rowsum,
    float* __restrict__ colsum, int Nv) {
  const int i = blockIdx.x, j = threadIdx.x;
  const int lane = j & 31, wid = j >> 5;
  const float l0 = a[i * 2 + 0] + c[j * 2 + 0];
  const float l1 = a[i * 2 + 1] + c[j * 2 + 1];
  const float m  = fmaxf(l0, l1);
  const float e0 = expf(l0 - m), e1 = expf(l1 - m);
  const float inv = 1.0f / (e0 + e1);
  const float p0 = e0 * inv, p1 = e1 * inv;
  const size_t o = ((size_t)i * Nv + j) * 2;
  out[o] = p0; out[o + 1] = p1;
  // log_softmax of probs
  const float m2  = fmaxf(p0, p1);
  const float lse = m2 + logf(expf(p0 - m2) + expf(p1 - m2));
  const float cell = -((gt[(size_t)i * Nv + j] != 0 ? p1 : p0) - lse);
  atomicAdd(&colsum[j], p1);
  __shared__ float sc[12], sp[12];
  const float cs = wave_sum(cell);
  const float ps = wave_sum(p1);
  if (lane == 0) { sc[wid] = cs; sp[wid] = ps; }
  __syncthreads();
  if (j == 0) {
    float t = 0, u = 0;
#pragma unroll
    for (int k = 0; k < 12; ++k) { t += sc[k]; u += sp[k]; }
    atomicAdd(cls_sum, t);
    rowsum[i] = u;
  }
}

__global__ __launch_bounds__(384) void finalize_k(
    const float* __restrict__ cls_sum, const float* __restrict__ rowsum,
    const float* __restrict__ colsum, float* __restrict__ tail, int Nv) {
  const int i = threadIdx.x, lane = i & 31, wid = i >> 5;
  float v = (i < Nv - 1) ? fabsf(rowsum[i] + colsum[i] - 1.0f) : 0.0f;
  __shared__ float s[12];
  v = wave_sum(v);
  if (lane == 0) s[wid] = v;
  __syncthreads();
  if (i == 0) {
    float t = 0;
#pragma unroll
    for (int k = 0; k < 12; ++k) t += s[k];
    tail[0] = cls_sum[0] / (float)Nv;  // sum_i mean_j = total / N
    tail[1] = t;
  }
}

__global__ void zero_k(float* p, int n) {
  const int i = blockIdx.x * 256 + threadIdx.x;
  if (i < n) p[i] = 0.0f;
}

// ---------------------------------------------------------------------------
// Host-side orchestration
// ---------------------------------------------------------------------------
struct EncP {
  const float *qkv_w, *qkv_b, *out_w, *out_b, *ln1_s, *ln1_b;
  const float *ff1_w, *ff1_b, *ff2_w, *ff2_b, *ln2_s, *ln2_b;
};

// Five specializations cover every call site:
//   TB=1,B=1,R=0,RELU=0 : qkv proj            TB=1,B=0,R=0,RELU=0 : Q@K^T
//   TB=0,B=0,R=0,RELU=0 : attn@V              TB=1,B=1,R=1,RELU=0 : out/ff2 (+residual)
//   TB=1,B=1,R=0,RELU=1 : ff1 (+relu)
static inline void launch_gemm(bool transB, bool hasBias, bool hasRes, bool relu,
                               const float* A, int lda, long long bsA,
                               const float* B, int ldb, long long bsB,
                               float* C, int ldc, long long bsC,
                               const float* bias, const float* res, int ldr,
                               int M, int Ncols, int K, int batch, hipStream_t s) {
  dim3 g(Ncols / TILE_N, M / TILE_M, batch), b(128);
  if (!transB)
    gemm_wmma<0, 0, 0, 0><<<g, b, 0, s>>>(A, lda, bsA, B, ldb, bsB, C, ldc, bsC,
                                          bias, res, ldr, M, Ncols, K);
  else if (hasRes)
    gemm_wmma<1, 1, 1, 0><<<g, b, 0, s>>>(A, lda, bsA, B, ldb, bsB, C, ldc, bsC,
                                          bias, res, ldr, M, Ncols, K);
  else if (relu)
    gemm_wmma<1, 1, 0, 1><<<g, b, 0, s>>>(A, lda, bsA, B, ldb, bsB, C, ldc, bsC,
                                          bias, res, ldr, M, Ncols, K);
  else if (hasBias)
    gemm_wmma<1, 1, 0, 0><<<g, b, 0, s>>>(A, lda, bsA, B, ldb, bsB, C, ldc, bsC,
                                          bias, res, ldr, M, Ncols, K);
  else
    gemm_wmma<1, 0, 0, 0><<<g, b, 0, s>>>(A, lda, bsA, B, ldb, bsB, C, ldc, bsC,
                                          bias, res, ldr, M, Ncols, K);
}

static void run_encoder(float* x, int d, const EncP& p, float* qkvbuf, float* scores,
                        float* obuf, float* hbuf, float* tmp, hipStream_t s) {
  const int Nt = 384, Hh = 4, Ll = 2, FFd = 2048;
  const int hd = d / Hh;
  const float scale = 1.0f / sqrtf((float)hd);
  for (int l = 0; l < Ll; ++l) {
    const float* qkvW = p.qkv_w + (size_t)l * 3 * d * d;
    const float* qkvB = p.qkv_b + (size_t)l * 3 * d;
    const float* outW = p.out_w + (size_t)l * d * d;
    const float* outB = p.out_b + (size_t)l * d;
    const float* ff1W = p.ff1_w + (size_t)l * FFd * d;
    const float* ff1B = p.ff1_b + (size_t)l * FFd;
    const float* ff2W = p.ff2_w + (size_t)l * d * FFd;
    const float* ff2B = p.ff2_b + (size_t)l * d;

    // qkv = x @ qkv_w^T + qkv_b
    launch_gemm(true, true, false, false, x, d, 0, qkvW, d, 0, qkvbuf, 3 * d, 0,
                qkvB, nullptr, 0, Nt, 3 * d, d, 1, s);
    // scores[h] = Q_h @ K_h^T  (batched over heads via blockIdx.z)
    launch_gemm(true, false, false, false, qkvbuf, 3 * d, hd, qkvbuf + d, 3 * d, hd,
                scores, Nt, (long long)Nt * Nt, nullptr, nullptr, 0, Nt, Nt, hd, Hh, s);
    softmax_rows<<<dim3(Hh * Nt), dim3(128), 0, s>>>(scores, Nt, scale);
    // O_h = attn_h @ V_h  (batched over heads)
    launch_gemm(false, false, false, false, scores, Nt, (long long)Nt * Nt,
                qkvbuf + 2 * d, 3 * d, hd, obuf, d, hd, nullptr, nullptr, 0,
                Nt, hd, Nt, Hh, s);
    // out proj + residual, then LN1
    launch_gemm(true, true, true, false, obuf, d, 0, outW, d, 0, tmp, d, 0,
                outB, x, d, Nt, d, d, 1, s);
    layernorm_k<<<Nt, 256, 0, s>>>(tmp, x, p.ln1_s + (size_t)l * d, p.ln1_b + (size_t)l * d, d);
    // FFN + residual, then LN2
    launch_gemm(true, true, false, true, x, d, 0, ff1W, d, 0, hbuf, FFd, 0,
                ff1B, nullptr, 0, Nt, FFd, d, 1, s);
    launch_gemm(true, true, true, false, hbuf, FFd, 0, ff2W, FFd, 0, tmp, d, 0,
                ff2B, x, d, Nt, d, FFd, 1, s);
    layernorm_k<<<Nt, 256, 0, s>>>(tmp, x, p.ln2_s + (size_t)l * d, p.ln2_b + (size_t)l * d, d);
  }
}

extern "C" void kernel_launch(void* const* d_in, const int* in_sizes, int n_in,
                              void* d_out, int out_size, void* d_ws, size_t ws_size,
                              hipStream_t stream) {
  (void)in_sizes; (void)n_in; (void)out_size; (void)ws_size;
  const int Nt = 384, TDv = 768, VDv = 768, CDv = 1536;

  const float* text = (const float*)d_in[0];
  const float* vis  = (const float*)d_in[1];
  const int*   gt   = (const int*)d_in[2];
  EncP tp = { (const float*)d_in[3],  (const float*)d_in[4],  (const float*)d_in[5],
              (const float*)d_in[6],  (const float*)d_in[7],  (const float*)d_in[8],
              (const float*)d_in[9],  (const float*)d_in[10], (const float*)d_in[11],
              (const float*)d_in[12], (const float*)d_in[13], (const float*)d_in[14] };
  EncP cp = { (const float*)d_in[15], (const float*)d_in[16], (const float*)d_in[17],
              (const float*)d_in[18], (const float*)d_in[19], (const float*)d_in[20],
              (const float*)d_in[21], (const float*)d_in[22], (const float*)d_in[23],
              (const float*)d_in[24], (const float*)d_in[25], (const float*)d_in[26] };
  const float* lin_w = (const float*)d_in[27];
  const float* lin_b = (const float*)d_in[28];
  float* outp = (float*)d_out;

  // workspace layout (floats)
  float* ws = (float*)d_ws;
  size_t o = 0;
  float* qkvbuf = ws + o; o += (size_t)Nt * 3 * CDv;      // 384 x 4608
  float* scores = ws + o; o += (size_t)4 * Nt * Nt;       // 4 x 384 x 384
  float* obuf   = ws + o; o += (size_t)Nt * CDv;          // 384 x 1536
  float* hbuf   = ws + o; o += (size_t)Nt * 2048;         // 384 x 2048
  float* tmp    = ws + o; o += (size_t)Nt * CDv;          // 384 x 1536
  float* xT     = ws + o; o += (size_t)Nt * TDv;          // 384 x 768
  float* xC     = ws + o; o += (size_t)Nt * CDv;          // 384 x 1536
  float* abuf   = ws + o; o += (size_t)Nt * 2;
  float* cbuf   = ws + o; o += (size_t)Nt * 2;
  float* cls    = ws + o; o += 1;
  float* colsum = ws + o; o += Nt;                        // contiguous with cls
  float* rowsum = ws + o; o += Nt;

  // text encoder (in-place on xT)
  hipMemcpyAsync(xT, text, (size_t)Nt * TDv * sizeof(float), hipMemcpyDeviceToDevice,
                 stream);
  run_encoder(xT, TDv, tp, qkvbuf, scores, obuf, hbuf, tmp, stream);

  // concat + cross encoder (in-place on xC)
  concat_k<<<Nt, 256, 0, stream>>>(xT, vis, xC, TDv, VDv);
  run_encoder(xC, CDv, cp, qkvbuf, scores, obuf, hbuf, tmp, stream);

  // head + pairwise probs + losses
  head_ac_k<<<Nt, 256, 0, stream>>>(xC, lin_w, lin_b, abuf, cbuf, CDv);
  zero_k<<<2, 256, 0, stream>>>(cls, 1 + Nt);  // cls_sum + colsum
  probs_k<<<Nt, Nt, 0, stream>>>(abuf, cbuf, gt, outp, cls, rowsum, colsum, Nt);
  finalize_k<<<1, Nt, 0, stream>>>(cls, rowsum, colsum, outp + (size_t)Nt * Nt * 2, Nt);
}
